// GNNTCN_76441827934507
// MI455X (gfx1250) — compile-verified, matrix operand order checked by reading, not run on hardware
//
#include <hip/hip_runtime.h>
#include <hip/hip_bf16.h>
#include <math.h>

// ---------------------------------------------------------------------------
// Types for CDNA5 WMMA (wave32): v_wmma_f32_16x16x32_bf16
// ---------------------------------------------------------------------------
typedef __bf16 bf16_t;
typedef __attribute__((ext_vector_type(16))) __bf16 v16bf;
typedef __attribute__((ext_vector_type(8)))  float  v8f;
typedef __attribute__((ext_vector_type(4)))  int    v4i;

// ISA 7.12.2 16-bit A/B layout: element e of the v16bf, lane-half hi (=lane/16)
// holds K = e + 8*((e>>3) + hi). Even/odd element pairs are consecutive K,
// which lets the compiler vectorize fragment fills into ds_load_b128 /
// global_load_b128 (confirmed in disassembly).
__device__ __forceinline__ int wmma_k(int e, int hi) { return e + 8 * ((e >> 3) + hi); }

__device__ __forceinline__ float gelu_f(float x) {
    return 0.5f * x * (1.0f + erff(x * 0.70710678118654752f));   // exact GELU
}

// CDNA5 async global->LDS DMA (ASYNCcnt-tracked), guarded so compilation
// falls back to plain loads on toolchains that do not declare the builtins.
// Probe result (round 3): builtin exists here; signature is
//   (v4i addrspace(1)* src, v4i addrspace(3)* dst, imm offset, imm cpol)
#if defined(__AMDGCN__) && __has_builtin(__builtin_amdgcn_global_load_async_to_lds_b128) && __has_builtin(__builtin_amdgcn_s_wait_asynccnt)
#define ASYNC_LDS 1
#else
#define ASYNC_LDS 0
#endif

#if ASYNC_LDS
__device__ __forceinline__ void async_copy16(void* lds_dst, const void* gsrc) {
    __builtin_amdgcn_global_load_async_to_lds_b128(
        (__attribute__((address_space(1))) v4i*)gsrc,
        (__attribute__((address_space(3))) v4i*)lds_dst,
        0, 0);
}
#endif

// Problem constants
#define B_   64
#define N_   40
#define W_   256
#define FIN_ 64
#define H_   4
#define D_   32
#define C_   128
#define L_   4
#define KW_  3
#define FC_  128
#define MROWS_ (B_ * N_ * W_)          // 655360 rows of C channels
#define KCONV_ (C_ * KW_)              // 384

// ---------------------------------------------------------------------------
// Kernel 1: repack weights to bf16 GEMM layouts
//   conv:  wq[(l*2+which)*128 + o][tap*128 + i]  <- c{1,2}w[l][o][i][tap]
//   gat:   wq_gat[c*64 + f]                      <- gat_w[c][f]
// ---------------------------------------------------------------------------
__global__ void prep_weights_kernel(const float* __restrict__ c1w,
                                    const float* __restrict__ c2w,
                                    const float* __restrict__ gat_w,
                                    bf16_t* __restrict__ wq_conv,
                                    bf16_t* __restrict__ wq_gat) {
    const int NCONV = L_ * 2 * C_ * KCONV_;   // 393216
    int idx = blockIdx.x * blockDim.x + threadIdx.x;
    if (idx < NCONV) {
        int lw   = idx / (C_ * KCONV_);
        int rest = idx % (C_ * KCONV_);
        int o    = rest / KCONV_;
        int kq   = rest % KCONV_;
        int tap  = kq >> 7;
        int i    = kq & 127;
        int l    = lw >> 1;
        const float* src = (lw & 1) ? c2w : c1w;
        wq_conv[idx] = (bf16_t)src[(((size_t)l * C_ + o) * C_ + i) * KW_ + tap];
    } else if (idx < NCONV + C_ * FIN_) {
        int j = idx - NCONV;
        wq_gat[j] = (bf16_t)gat_w[j];
    }
}

// ---------------------------------------------------------------------------
// Kernel 2: GAT projection h = x @ gat_w^T  (WMMA bf16, fp32 accum)
// fused with e_src/e_dst per-head reductions; h stored coalesced via LDS.
// ---------------------------------------------------------------------------
__global__ void gat_project_kernel(const float* __restrict__ x,
                                   const bf16_t* __restrict__ wq,
                                   const float* __restrict__ a_src,
                                   const float* __restrict__ a_dst,
                                   float* __restrict__ h_out,
                                   float* __restrict__ e_src,
                                   float* __restrict__ e_dst) {
    __shared__ bf16_t xs[16][FIN_];
    __shared__ float  hs[16][C_];
    const int tid  = threadIdx.x;
    const size_t row0 = (size_t)blockIdx.x * 16;

    for (int i = tid; i < 16 * FIN_; i += 256) {
        int ri = i >> 6, f = i & 63;
        xs[ri][f] = (bf16_t)x[(row0 + ri) * FIN_ + f];
    }
    __syncthreads();

    const int wv = tid >> 5, lane = tid & 31;
    const int hi = lane >> 4, ln = lane & 15;
    const int c0 = wv * 16;

    v8f acc = {};
#pragma unroll
    for (int kk = 0; kk < 2; ++kk) {
        v16bf a, b;
#pragma unroll
        for (int e = 0; e < 16; ++e) {
            int kq = kk * 32 + wmma_k(e, hi);
            a[e] = xs[ln][kq];
            b[e] = wq[(size_t)(c0 + ln) * FIN_ + kq];
        }
        acc = __builtin_amdgcn_wmma_f32_16x16x32_bf16(false, a, false, b,
                                                      (short)0, acc, false, false);
    }
#pragma unroll
    for (int q = 0; q < 8; ++q)
        hs[q + 8 * hi][c0 + ln] = acc[q];
    __syncthreads();

    // coalesced h store from the LDS tile
    for (int i = tid; i < 16 * C_; i += 256)
        h_out[(row0 + (i >> 7)) * C_ + (i & 127)] = hs[i >> 7][i & 127];

    if (tid < 64) {                       // 16 rows x 4 heads
        int ri = tid >> 2, hh = tid & 3;
        float ss = 0.f, sd = 0.f;
#pragma unroll
        for (int d = 0; d < D_; ++d) {
            float v = hs[ri][hh * D_ + d];
            ss += v * a_src[hh * D_ + d];
            sd += v * a_dst[hh * D_ + d];
        }
        e_src[(row0 + ri) * H_ + hh] = ss;
        e_dst[(row0 + ri) * H_ + hh] = sd;
    }
}

// ---------------------------------------------------------------------------
// Kernel 3: attention scores + softmax + aggregation. One block per (b,w,head).
// ---------------------------------------------------------------------------
__global__ void attn_kernel(const float* __restrict__ e_src,
                            const float* __restrict__ e_dst,
                            const float* __restrict__ adj,
                            const float* __restrict__ h,
                            float* __restrict__ gat) {
    __shared__ float es[N_], ed[N_];
    __shared__ float sm[N_][N_];
    __shared__ float hb[N_][D_];
    int blk = blockIdx.x;
    int hh  = blk & 3;
    int w   = (blk >> 2) & 255;
    int b   = blk >> 10;
    int tid = threadIdx.x;   // 128

    for (int j = tid; j < N_; j += 128) {
        size_t r = (size_t)(b * N_ + j) * W_ + w;
        es[j] = e_src[r * H_ + hh];
        ed[j] = e_dst[r * H_ + hh];
    }
    for (int i = tid; i < N_ * D_; i += 128) {
        int j = i >> 5, d = i & 31;
        hb[j][d] = h[((size_t)(b * N_ + j) * W_ + w) * C_ + hh * D_ + d];
    }
    __syncthreads();

    for (int i = tid; i < N_ * N_; i += 128) {
        int n = i / N_, j = i % N_;
        float sc = es[n] + ed[j];
        sc = sc > 0.f ? sc : 0.2f * sc;              // leaky_relu(0.2)
        sm[n][j] = (adj[n * N_ + j] != 0.f) ? sc : -1e9f;
    }
    __syncthreads();

    if (tid < N_) {
        float mx = -3.4e38f;
        for (int j = 0; j < N_; ++j) mx = fmaxf(mx, sm[tid][j]);
        float s = 0.f;
        for (int j = 0; j < N_; ++j) { float e = __expf(sm[tid][j] - mx); sm[tid][j] = e; s += e; }
        float inv = 1.f / s;
        for (int j = 0; j < N_; ++j) sm[tid][j] *= inv;
    }
    __syncthreads();

    for (int i = tid; i < N_ * D_; i += 128) {
        int n = i >> 5, d = i & 31;
        float acc = 0.f;
        for (int j = 0; j < N_; ++j) acc += sm[n][j] * hb[j][d];
        gat[((size_t)(b * N_ + n) * W_ + w) * C_ + hh * D_ + d] = acc;
    }
}

// ---------------------------------------------------------------------------
// Kernel 4: LayerNorm over C, row-wise (layout preserved: no transpose needed
// since activations stay (row, C) = (B*N*W, 128)). Tile staged via async DMA.
// ---------------------------------------------------------------------------
__global__ void ln_kernel(const float* __restrict__ gat,
                          const float* __restrict__ ln_g,
                          const float* __restrict__ ln_b,
                          float* __restrict__ seq) {
    __shared__ float tile[32][C_];
    __shared__ float mu[32], rs[32];
    const size_t row0 = (size_t)blockIdx.x * 32;
    int tid = threadIdx.x;  // 256

#if ASYNC_LDS
    for (int i = tid; i < 32 * 32; i += 256) {           // 16B segments
        int wl = i >> 5, seg = i & 31;
        async_copy16(&tile[wl][seg * 4], &gat[(row0 + wl) * C_ + seg * 4]);
    }
    __builtin_amdgcn_s_wait_asynccnt(0);
#else
    for (int i = tid; i < 32 * C_; i += 256) {
        int wl = i >> 7, c = i & 127;
        tile[wl][c] = gat[(row0 + wl) * C_ + c];
    }
#endif
    __syncthreads();

    if (tid < 32) {
        float s = 0.f;
        for (int c = 0; c < C_; ++c) s += tile[tid][c];
        float m = s * (1.f / C_);
        float v = 0.f;
        for (int c = 0; c < C_; ++c) { float d = tile[tid][c] - m; v += d * d; }
        mu[tid] = m;
        rs[tid] = rsqrtf(v * (1.f / C_) + 1e-5f);
    }
    __syncthreads();
    for (int i = tid; i < 32 * C_; i += 256) {
        int wl = i >> 7, c = i & 127;
        float val = (tile[wl][c] - mu[wl]) * rs[wl] * ln_g[c] + ln_b[c];
        seq[(row0 + wl) * C_ + c] = val;
    }
}

// ---------------------------------------------------------------------------
// Kernel 5: dilated causal conv as implicit GEMM (WMMA), fused epilogue.
// Activations in (s, W, C): LDS staging is a layout-preserving row copy, so
// the bf16 input path uses GLOBAL_LOAD_ASYNC_TO_LDS_B128 (ASYNCcnt DMA).
// B-fragments (16x384 weight slice per wave) hoisted into resident VGPRs.
// ---------------------------------------------------------------------------
template <typename TIn, typename TOut, bool HAS_RES>
__global__ void tcn_conv_kernel(const TIn* __restrict__ in,
                                TOut* __restrict__ out,
                                const bf16_t* __restrict__ wq,   // [128][384]
                                const float* __restrict__ cb,    // bias  [C]
                                const float* __restrict__ bns,
                                const float* __restrict__ bnb,
                                const float* __restrict__ bnm,
                                const float* __restrict__ bnv,
                                const float* __restrict__ res,   // fp32, HAS_RES
                                int dil) {
    __shared__ bf16_t zsh[144][C_];      // [time 16(pad)+128][channels]
    int blk   = blockIdx.x;
    int s     = blk >> 1;
    int cbase = (blk & 1) * 128;
    int tid   = threadIdx.x;

#if ASYNC_LDS
    if constexpr (sizeof(TIn) == 2) {
        if (cbase == 0) {                 // causal zero pad, chunk 0 only
            for (int i = tid; i < 16 * C_; i += 256)
                zsh[i >> 7][i & 127] = (bf16_t)0.f;
        }
        for (int i = tid; i < 144 * (C_ / 8); i += 256) {   // 16B (8 bf16) segs
            int tl = i >> 4, seg = i & 15;
            int tg = cbase - 16 + tl;
            if (tg >= 0)
                async_copy16(&zsh[tl][seg * 8], &in[((size_t)s * W_ + tg) * C_ + seg * 8]);
        }
        __builtin_amdgcn_s_wait_asynccnt(0);
    } else
#endif
    {
        for (int i = tid; i < 144 * C_; i += 256) {
            int tl = i >> 7, ci = i & 127;
            int tg = cbase - 16 + tl;
            zsh[tl][ci] = (tg >= 0) ? (bf16_t)in[((size_t)s * W_ + tg) * C_ + ci]
                                    : (bf16_t)0.f;
        }
    }
    if (HAS_RES) {
        // Warm caches for the residual tile (emits global_prefetch_b8).
        for (int i = tid; i < 256; i += 256)
            __builtin_prefetch(&res[((size_t)s * W_ + cbase) * C_ + i * 64], 0, 1);
    }
    __syncthreads();

    const int wv = tid >> 5, lane = tid & 31;
    const int hi = lane >> 4, ln = lane & 15;
    const int c0 = wv * 16;
    const int o  = c0 + ln;              // this lane's output channel column

    float scale = bns[o] * rsqrtf(bnv[o] + 1e-5f);
    float shift = bnb[o] - bnm[o] * scale;
    float bias  = cb[o];

    // Hoist all 12 B-fragments (wave's 16x384 weight slice) into registers.
    v16bf bfrag[12];
#pragma unroll
    for (int kk = 0; kk < 12; ++kk) {
#pragma unroll
        for (int e = 0; e < 16; ++e)
            bfrag[kk][e] = wq[(size_t)o * KCONV_ + kk * 32 + wmma_k(e, hi)];
    }

    for (int tt = 0; tt < 8; ++tt) {     // 8 time tiles of 16 per chunk
        v8f acc = {};
#pragma unroll
        for (int kk = 0; kk < 12; ++kk) {
            int tap  = (kk * 32) >> 7;
            int ib   = (kk * 32) & 127;
            int trow = tt * 16 + ln + 16 - (2 - tap) * dil;  // in [0,144)
            v16bf a;
#pragma unroll
            for (int e = 0; e < 16; ++e)
                a[e] = zsh[trow][ib + wmma_k(e, hi)];
            acc = __builtin_amdgcn_wmma_f32_16x16x32_bf16(false, a, false, bfrag[kk],
                                                          (short)0, acc, false, false);
        }
#pragma unroll
        for (int q = 0; q < 8; ++q) {
            int t = cbase + tt * 16 + q + 8 * hi;
            float y = gelu_f((acc[q] + bias) * scale + shift);
            size_t oi = ((size_t)s * W_ + t) * C_ + o;
            if (HAS_RES) y = gelu_f(y + res[oi]);
            out[oi] = (TOut)y;
        }
    }
}

// ---------------------------------------------------------------------------
// Kernel 6: final MLP head. One block per (b, n<7).
// ---------------------------------------------------------------------------
__global__ void head_kernel(const float* __restrict__ seq,
                            const float* __restrict__ f1w,
                            const float* __restrict__ f1b,
                            const float* __restrict__ f2w,
                            const float* __restrict__ f2b,
                            float* __restrict__ outp) {
    __shared__ float tl[C_];
    __shared__ float red[FC_];
    int blk = blockIdx.x;               // b*7 + n
    int b = blk / 7, n = blk % 7;
    int bn = b * N_ + n;
    int tid = threadIdx.x;              // 128

    tl[tid] = seq[((size_t)bn * W_ + (W_ - 1)) * C_ + tid];  // contiguous row
    __syncthreads();
    float acc = f1b[tid];
    for (int c = 0; c < C_; ++c) acc += tl[c] * f1w[tid * C_ + c];
    red[tid] = gelu_f(acc) * f2w[tid];
    __syncthreads();
    for (int st = 64; st > 0; st >>= 1) {
        if (tid < st) red[tid] += red[tid + st];
        __syncthreads();
    }
    if (tid == 0) outp[blk] = red[0] + f2b[0];
}

// ---------------------------------------------------------------------------
// Launch
// ---------------------------------------------------------------------------
extern "C" void kernel_launch(void* const* d_in, const int* in_sizes, int n_in,
                              void* d_out, int out_size, void* d_ws, size_t ws_size,
                              hipStream_t stream) {
    const float* x     = (const float*)d_in[0];
    const float* adj   = (const float*)d_in[1];
    const float* gat_w = (const float*)d_in[2];
    const float* a_src = (const float*)d_in[3];
    const float* a_dst = (const float*)d_in[4];
    const float* ln_g  = (const float*)d_in[5];
    const float* ln_b  = (const float*)d_in[6];
    const float* c1w   = (const float*)d_in[7];
    const float* c1b   = (const float*)d_in[8];
    const float* c2w   = (const float*)d_in[9];
    const float* c2b   = (const float*)d_in[10];
    const float* bn1s  = (const float*)d_in[11];
    const float* bn1b  = (const float*)d_in[12];
    const float* bn1m  = (const float*)d_in[13];
    const float* bn1v  = (const float*)d_in[14];
    const float* bn2s  = (const float*)d_in[15];
    const float* bn2b  = (const float*)d_in[16];
    const float* bn2m  = (const float*)d_in[17];
    const float* bn2v  = (const float*)d_in[18];
    const float* f1w   = (const float*)d_in[19];
    const float* f1b   = (const float*)d_in[20];
    const float* f2w   = (const float*)d_in[21];
    const float* f2b   = (const float*)d_in[22];

    char* ws = (char*)d_ws;
    size_t off = 0;
    auto take = [&](size_t bytes) -> char* {
        char* p = ws + off;
        off = (off + bytes + 255) & ~(size_t)255;
        return p;
    };
    bf16_t* wq_conv = (bf16_t*)take((size_t)L_ * 2 * C_ * KCONV_ * sizeof(bf16_t));
    bf16_t* wq_gat  = (bf16_t*)take((size_t)C_ * FIN_ * sizeof(bf16_t));
    float*  e_src   = (float*) take((size_t)MROWS_ * H_ * sizeof(float));
    float*  e_dst   = (float*) take((size_t)MROWS_ * H_ * sizeof(float));
    float*  buf_h   = (float*) take((size_t)MROWS_ * C_ * sizeof(float));
    float*  buf_a   = (float*) take((size_t)MROWS_ * C_ * sizeof(float));
    float*  buf_s   = (float*) take((size_t)MROWS_ * C_ * sizeof(float));
    // bf16 conv1->conv2 temp aliases the (then-dead) h buffer.
    bf16_t* buf_t   = (bf16_t*)buf_h;

    prep_weights_kernel<<<(L_ * 2 * C_ * KCONV_ + C_ * FIN_ + 255) / 256, 256, 0, stream>>>(
        c1w, c2w, gat_w, wq_conv, wq_gat);

    gat_project_kernel<<<MROWS_ / 16, 256, 0, stream>>>(
        x, wq_gat, a_src, a_dst, buf_h, e_src, e_dst);

    attn_kernel<<<B_ * W_ * H_, 128, 0, stream>>>(e_src, e_dst, adj, buf_h, buf_a);

    ln_kernel<<<MROWS_ / 32, 256, 0, stream>>>(buf_a, ln_g, ln_b, buf_s);

    for (int l = 0; l < L_; ++l) {
        int dil = 1 << l;
        tcn_conv_kernel<float, bf16_t, false><<<B_ * N_ * 2, 256, 0, stream>>>(
            buf_s, buf_t, wq_conv + (size_t)(l * 2 + 0) * C_ * KCONV_,
            c1b + l * C_, bn1s + l * C_, bn1b + l * C_, bn1m + l * C_, bn1v + l * C_,
            nullptr, dil);
        tcn_conv_kernel<bf16_t, float, true><<<B_ * N_ * 2, 256, 0, stream>>>(
            buf_t, buf_s, wq_conv + (size_t)(l * 2 + 1) * C_ * KCONV_,
            c2b + l * C_, bn2s + l * C_, bn2b + l * C_, bn2m + l * C_, bn2v + l * C_,
            buf_s, dil);   // residual read + in-place write (elementwise-safe)
    }

    head_kernel<<<B_ * 7, 128, 0, stream>>>(buf_s, f1w, f1b, f2w, f2b, (float*)d_out);
}
